// Decoder_5669356830763
// MI455X (gfx1250) — matrix operand order repaired and checked by reading, not compiled, not run on hardware
//
#include <hip/hip_runtime.h>
#include <hip/hip_bf16.h>
#include <math.h>

// Problem constants (from reference)
#define S_ 512
#define B_ 64
#define E_ 2048
#define H_ 512
#define V_ 32000

typedef __attribute__((ext_vector_type(16))) __bf16 bf16x16;
typedef __attribute__((ext_vector_type(8)))  __bf16 bf16x8;
typedef __attribute__((ext_vector_type(8)))  float  f32x8;

__device__ __forceinline__ bf16x8 cvt8(const float* __restrict__ p) {
    f32x8 v = *(const f32x8*)p;
    bf16x8 r;
#pragma unroll
    for (int i = 0; i < 8; ++i) r[i] = (__bf16)v[i];
    return r;
}

__device__ __forceinline__ float sigmoidf_(float x) { return 1.0f / (1.0f + __expf(-x)); }

// ---------------------------------------------------------------------------
// Generic WMMA GEMM: C[M,N] = A[M,K] @ W[N,K]^T + bias[N]  (+= if ACC)
// One wave32 computes one 16x16 C tile; K in steps of 32 (bf16 on the fly).
// ---------------------------------------------------------------------------
template <int ACC>
__global__ __launch_bounds__(256) void gemm_wmma_bf16(
    const float* __restrict__ A, const float* __restrict__ W,
    const float* __restrict__ bias, float* __restrict__ C,
    int M, int N, int K, int lda, int ldw, int ldc)
{
    int gwave = (blockIdx.x * blockDim.x + threadIdx.x) >> 5;
    int lane  = threadIdx.x & 31;
    int ntiles = N >> 4;
    int total  = (M >> 4) * ntiles;
    if (gwave >= total) return;

    int tm = (gwave / ntiles) << 4;
    int tn = (gwave % ntiles) << 4;

    int ml   = lane & 15;      // A row within tile / B column within tile
    int half = lane >> 4;      // which half-wave (K split)

    const float* arow = A + (long)(tm + ml) * lda;
    const float* wrow = W + (long)(tn + ml) * ldw;
    int kA = half * 8;         // A: lanes 0-15 -> K 0..7 & 16..23; 16-31 -> 8..15 & 24..31
    int kB = half * 8;         // B: lanes 0-15 -> K 0..7 & 16..23; 16-31 -> 8..15 & 24..31

    f32x8 acc = {};
    for (int k0 = 0; k0 < K; k0 += 32) {
        bf16x8 a0 = cvt8(arow + k0 + kA);
        bf16x8 a1 = cvt8(arow + k0 + 16 + kA);
        bf16x8 b0 = cvt8(wrow + k0 + kB);
        bf16x8 b1 = cvt8(wrow + k0 + kB + 16);
        bf16x16 af, bfm;
#pragma unroll
        for (int i = 0; i < 8; ++i) {
            af[i] = a0[i];  af[8 + i] = a1[i];
            bfm[i] = b0[i]; bfm[8 + i] = b1[i];
        }
        acc = __builtin_amdgcn_wmma_f32_16x16x32_bf16(
            false, af, false, bfm, (short)0, acc, false, false);
    }

    int n     = tn + ml;
    int mbase = tm + half * 8;   // C: vgpr r -> row mbase+r, col = lane&15 (per half)
    float bv  = bias[n];
#pragma unroll
    for (int r = 0; r < 8; ++r) {
        float v = acc[r] + bv;
        long idx = (long)(mbase + r) * ldc + n;
        if (ACC) v += C[idx];
        C[idx] = v;
    }
}

// ---------------------------------------------------------------------------
// Fused Bahdanau energy + score:
//   scores[b,s] = sum_h tanh( enc[s,b,:] . w1[h,:] + biasc[b,h] + w1b[h] ) * attv[h]
// One workgroup (8 waves) per 16 rows of the [S*B, E] A matrix.
// A-tile (16x32 fp32, 2KB) is double-buffered in LDS via
// GLOBAL_LOAD_ASYNC_TO_LDS_B128 (ASYNCcnt-tracked, no VGPR staging):
// each wave's lanes 0..15 move one 16B chunk per batch -> 1 ASYNCcnt per wave.
// Each wave owns 4 N-tiles (8 waves * 4 * 16 = 512 = all of H); all 4 B
// fragments are loaded in one clause before the 4 back-to-back WMMAs.
// ---------------------------------------------------------------------------
__global__ __launch_bounds__(256) void energy_scores_kernel(
    const float*  __restrict__ enc,    // [S*B, E] row-major fp32
    const __bf16* __restrict__ w1bf,   // [H, E] bf16 (L2-resident)
    const float*  __restrict__ biasc,  // [B, H]  (w2_b + last_h @ w2^T)
    const float*  __restrict__ w1b,    // [H]
    const float*  __restrict__ attv,   // [H]
    float*        __restrict__ scores) // [B, S]
{
    __shared__ float atile[2][16 * 32];   // 2 x 2KB fp32 staging
    __shared__ float ssc[16];

    int tid  = threadIdx.x;
    int wave = tid >> 5;
    int lane = tid & 31;
    int tm   = blockIdx.x * 16;           // global row base into [S*B, E]

    if (tid < 16) ssc[tid] = 0.0f;

    int ml   = lane & 15;
    int half = lane >> 4;
    int kA   = half * 8;
    int kB   = half * 8;

    // async loader: 8 waves x lanes 0..15 = 128 chunks of 16B = 2KB tile
    bool loader = (lane < 16);
    int  chunk  = wave * 16 + lane;                    // 0..127
    int  lrow   = chunk >> 3;                          // 8 x 16B chunks per 128B row
    int  lseg   = chunk & 7;
    const float* gsrc = enc + (size_t)(tm + lrow) * E_ + lseg * 4;
    unsigned lds0 = (unsigned)(size_t)(&atile[0][0]) + chunk * 16;
    unsigned lds1 = (unsigned)(size_t)(&atile[1][0]) + chunk * 16;

    if (loader) {
        unsigned long long ga = (unsigned long long)(size_t)gsrc;
        asm volatile("global_load_async_to_lds_b128 %0, %1, off"
                     :: "v"(lds0), "v"(ga) : "memory");
    }

    f32x8 acc[4] = {};

    for (int k0 = 0; k0 < E_; k0 += 32) {
        int cur = (k0 >> 5) & 1;
        if (k0 + 32 < E_) {
            // prefetch next K-block into the other buffer (its previous readers
            // finished at the barrier that ended the previous iteration)
            if (loader) {
                unsigned long long ga = (unsigned long long)(size_t)(gsrc + k0 + 32);
                unsigned dst = cur ? lds0 : lds1;
                asm volatile("global_load_async_to_lds_b128 %0, %1, off"
                             :: "v"(dst), "v"(ga) : "memory");
            }
            // oldest batch (current buffer) has landed when <=1 remains in flight
            asm volatile("s_wait_asynccnt 1" ::: "memory");
        } else {
            asm volatile("s_wait_asynccnt 0" ::: "memory");
        }
        __syncthreads();   // all waves' chunks of the current buffer are visible

        // B fragments for all 4 N-tiles first (one clause of global_load_b128s)
        bf16x16 bfrag[4];
#pragma unroll
        for (int t = 0; t < 4; ++t) {
            const __bf16* wrow = w1bf + (long)((wave * 4 + t) * 16 + ml) * E_ + k0 + kB;
            bf16x8 b0 = *(const bf16x8*)(wrow);
            bf16x8 b1 = *(const bf16x8*)(wrow + 16);
#pragma unroll
            for (int i = 0; i < 8; ++i) { bfrag[t][i] = b0[i]; bfrag[t][8 + i] = b1[i]; }
        }

        // A fragment from LDS (fp32 -> bf16 in regs)
        const float* at = atile[cur];
        bf16x16 af;
        {
            f32x8 va0 = *(const f32x8*)(at + ml * 32 + kA);
            f32x8 va1 = *(const f32x8*)(at + ml * 32 + 16 + kA);
#pragma unroll
            for (int i = 0; i < 8; ++i) {
                af[i]     = (__bf16)va0[i];
                af[8 + i] = (__bf16)va1[i];
            }
        }

        // 4 independent WMMAs back-to-back (no D->A/B hazard)
#pragma unroll
        for (int t = 0; t < 4; ++t)
            acc[t] = __builtin_amdgcn_wmma_f32_16x16x32_bf16(
                false, af, false, bfrag[t], (short)0, acc[t], false, false);

        __syncthreads();   // everyone done reading buf[cur]; next iter overwrites it
    }

    // epilogue: tanh + att_v dot, reduce per row via LDS atomics
    int mbase = half * 8;
    float partial[8];
#pragma unroll
    for (int r = 0; r < 8; ++r) partial[r] = 0.0f;

#pragma unroll
    for (int t = 0; t < 4; ++t) {
        int h  = (wave * 4 + t) * 16 + ml;
        float av = attv[h];
        float wb = w1b[h];
#pragma unroll
        for (int r = 0; r < 8; ++r) {
            int row = tm + mbase + r;       // row = s*B + b
            int b   = row & (B_ - 1);
            float e = tanhf(acc[t][r] + biasc[b * H_ + h] + wb);
            partial[r] += av * e;
        }
    }
#pragma unroll
    for (int r = 0; r < 8; ++r)
        atomicAdd(&ssc[mbase + r], partial[r]);
    __syncthreads();

    if (tid < 16) {
        int row = tm + tid;
        int b = row & (B_ - 1);
        int s = row >> 6;                   // / B_
        scores[b * S_ + s] = ssc[tid];
    }
}

// ---------------------------------------------------------------------------
// Streaming / elementwise kernels
// ---------------------------------------------------------------------------
__global__ void mean_kernel(const float* __restrict__ enc, float* __restrict__ mean) {
    int idx = blockIdx.x * blockDim.x + threadIdx.x;   // over B*E
    if (idx >= B_ * E_) return;
    float s = 0.0f;
    for (int t = 0; t < S_; ++t) s += enc[(long)t * B_ * E_ + idx];
    mean[idx] = s * (1.0f / S_);
}

__global__ void cvt_bf16_kernel(const float* __restrict__ src, __bf16* __restrict__ dst, int n) {
    int idx = blockIdx.x * blockDim.x + threadIdx.x;
    if (idx < n) dst[idx] = (__bf16)src[idx];
}

__global__ void context_kernel(const float* __restrict__ enc, const float* __restrict__ weight,
                               float* __restrict__ ctx) {
    int idx = blockIdx.x * blockDim.x + threadIdx.x;   // over B*E
    if (idx >= B_ * E_) return;
    int b = idx / E_;
    float acc = 0.0f;
    for (int t = 0; t < S_; ++t)
        acc += weight[b * S_ + t] * enc[(long)t * B_ * E_ + idx];
    ctx[idx] = acc;
}

__global__ void embed_kernel(const int* __restrict__ input_step, const float* __restrict__ emb,
                             float* __restrict__ out) {
    int idx = blockIdx.x * blockDim.x + threadIdx.x;   // over B*H
    if (idx >= B_ * H_) return;
    int b = idx / H_, h = idx - b * H_;
    out[idx] = emb[(long)input_step[b] * H_ + h];
}

__global__ void concat_kernel(const float* __restrict__ last_h, const float* __restrict__ ctx,
                              float* __restrict__ cat) {
    int idx = blockIdx.x * blockDim.x + threadIdx.x;   // over B*(H+E)
    const int W = H_ + E_;
    if (idx >= B_ * W) return;
    int b = idx / W, j = idx - b * W;
    cat[idx] = (j < H_) ? last_h[b * H_ + j] : ctx[b * E_ + (j - H_)];
}

__global__ void lstm_kernel(const float* __restrict__ gates, const float* __restrict__ cprev,
                            float* __restrict__ hout, float* __restrict__ cout) {
    int idx = blockIdx.x * blockDim.x + threadIdx.x;   // over B*H
    if (idx >= B_ * H_) return;
    int b = idx / H_, h = idx - b * H_;
    const float* g = gates + (long)b * 4 * H_;
    float ig = sigmoidf_(g[h]);
    float fg = sigmoidf_(g[H_ + h]);
    float gg = tanhf(g[2 * H_ + h]);
    float og = sigmoidf_(g[3 * H_ + h]);
    float c  = fg * cprev[idx] + ig * gg;
    cout[idx] = c;
    hout[idx] = og * tanhf(c);
}

// in-place row softmax: one block per row of length L (row stride == L)
__global__ __launch_bounds__(256) void softmax_rows_kernel(float* __restrict__ x, int L) {
    __shared__ float red[256];
    float* row = x + (long)blockIdx.x * L;
    int tid = threadIdx.x;

    float m = -3.402823e38f;
    for (int i = tid; i < L; i += 256) m = fmaxf(m, row[i]);
    red[tid] = m; __syncthreads();
    for (int s = 128; s > 0; s >>= 1) {
        if (tid < s) red[tid] = fmaxf(red[tid], red[tid + s]);
        __syncthreads();
    }
    m = red[0]; __syncthreads();

    float sum = 0.0f;
    for (int i = tid; i < L; i += 256) {
        float e = __expf(row[i] - m);
        row[i] = e;
        sum += e;
    }
    red[tid] = sum; __syncthreads();
    for (int s = 128; s > 0; s >>= 1) {
        if (tid < s) red[tid] += red[tid + s];
        __syncthreads();
    }
    float inv = 1.0f / red[0];
    for (int i = tid; i < L; i += 256) row[i] *= inv;
}

// ---------------------------------------------------------------------------
// Host-side orchestration
// ---------------------------------------------------------------------------
static inline void launch_gemm(const float* A, const float* W, const float* bias, float* C,
                               int M, int N, int K, int lda, int ldw, int ldc,
                               int accumulate, hipStream_t stream) {
    int waves  = (M / 16) * (N / 16);
    int blocks = (waves + 7) / 8;
    if (accumulate)
        gemm_wmma_bf16<1><<<blocks, 256, 0, stream>>>(A, W, bias, C, M, N, K, lda, ldw, ldc);
    else
        gemm_wmma_bf16<0><<<blocks, 256, 0, stream>>>(A, W, bias, C, M, N, K, lda, ldw, ldc);
}

extern "C" void kernel_launch(void* const* d_in, const int* in_sizes, int n_in,
                              void* d_out, int out_size, void* d_ws, size_t ws_size,
                              hipStream_t stream) {
    const int*   input_step = (const int*)  d_in[0];
    const float* enc        = (const float*)d_in[1];
    const float* emb        = (const float*)d_in[2];
    const float* att_v      = (const float*)d_in[3];
    const float* w1_w       = (const float*)d_in[4];
    const float* w1_b       = (const float*)d_in[5];
    const float* w2_w       = (const float*)d_in[6];
    const float* w2_b       = (const float*)d_in[7];
    const float* h_w        = (const float*)d_in[8];
    const float* h_b        = (const float*)d_in[9];
    const float* w_ih       = (const float*)d_in[10];
    const float* w_hh       = (const float*)d_in[11];
    const float* b_ih       = (const float*)d_in[12];
    const float* b_hh       = (const float*)d_in[13];
    const float* out_w      = (const float*)d_in[14];
    const float* out_b      = (const float*)d_in[15];
    const float* init_h_w   = (const float*)d_in[16];
    const float* init_h_b   = (const float*)d_in[17];
    const float* init_c_w   = (const float*)d_in[18];
    const float* init_c_b   = (const float*)d_in[19];

    // workspace layout
    char* base = (char*)d_ws;
    size_t off = 0;
    auto alloc = [&](size_t bytes) -> char* {
        char* p = base + off;
        off += (bytes + 255) & ~(size_t)255;
        return p;
    };
    float*  ws_mean  = (float*) alloc((size_t)B_ * E_ * 4);
    float*  ws_lh    = (float*) alloc((size_t)B_ * H_ * 4);
    float*  ws_lc    = (float*) alloc((size_t)B_ * H_ * 4);
    float*  ws_biasc = (float*) alloc((size_t)B_ * H_ * 4);
    float*  ws_sc    = (float*) alloc((size_t)B_ * S_ * 4);
    float*  ws_ctx   = (float*) alloc((size_t)B_ * E_ * 4);
    float*  ws_x     = (float*) alloc((size_t)B_ * H_ * 4);
    float*  ws_cat   = (float*) alloc((size_t)B_ * (E_ + H_) * 4);
    float*  ws_h0    = (float*) alloc((size_t)B_ * H_ * 4);
    float*  ws_gates = (float*) alloc((size_t)B_ * 4 * H_ * 4);
    __bf16* ws_w1bf  = (__bf16*)alloc((size_t)H_ * E_ * 2);

    float* out_probs = (float*)d_out;                         // [B, V]
    float* out_h     = out_probs + (size_t)B_ * V_;           // [B, H]
    float* out_c     = out_h + (size_t)B_ * H_;               // [B, H]

    // 1. mean over S
    mean_kernel<<<(B_ * E_ + 255) / 256, 256, 0, stream>>>(enc, ws_mean);

    // 2-3. last_h / last_c
    launch_gemm(ws_mean, init_h_w, init_h_b, ws_lh, B_, H_, E_, E_, E_, H_, 0, stream);
    launch_gemm(ws_mean, init_c_w, init_c_b, ws_lc, B_, H_, E_, E_, E_, H_, 0, stream);

    // 4. w1 -> bf16 (reused by 2048 workgroups; L2-resident)
    cvt_bf16_kernel<<<(H_ * E_ + 255) / 256, 256, 0, stream>>>(w1_w, ws_w1bf, H_ * E_);

    // 5. biasc = last_h @ w2^T + w2_b
    launch_gemm(ws_lh, w2_w, w2_b, ws_biasc, B_, H_, H_, H_, H_, H_, 0, stream);

    // 6. fused energy + scores (async double-buffered LDS staging)
    energy_scores_kernel<<<(S_ * B_) / 16, 256, 0, stream>>>(
        enc, ws_w1bf, ws_biasc, w1_b, att_v, ws_sc);

    // 7. softmax over S
    softmax_rows_kernel<<<B_, 256, 0, stream>>>(ws_sc, S_);

    // 8. context
    context_kernel<<<(B_ * E_ + 255) / 256, 256, 0, stream>>>(enc, ws_sc, ws_ctx);

    // 9. embedding gather
    embed_kernel<<<(B_ * H_ + 255) / 256, 256, 0, stream>>>(input_step, emb, ws_x);

    // 10. concat [last_h, context]
    concat_kernel<<<(B_ * (E_ + H_) + 255) / 256, 256, 0, stream>>>(ws_lh, ws_ctx, ws_cat);

    // 11. h0 = cat @ h_w^T + h_b
    launch_gemm(ws_cat, h_w, h_b, ws_h0, B_, H_, E_ + H_, E_ + H_, E_ + H_, H_, 0, stream);

    // 12-13. gates = x @ w_ih^T + b_ih + h0 @ w_hh^T + b_hh
    launch_gemm(ws_x,  w_ih, b_ih, ws_gates, B_, 4 * H_, H_, H_, H_, 4 * H_, 0, stream);
    launch_gemm(ws_h0, w_hh, b_hh, ws_gates, B_, 4 * H_, H_, H_, H_, 4 * H_, 1, stream);

    // 14. LSTM cell -> h, c outputs
    lstm_kernel<<<(B_ * H_ + 255) / 256, 256, 0, stream>>>(ws_gates, ws_lc, out_h, out_c);

    // 15. logits = h @ out_w^T + out_b
    launch_gemm(out_h, out_w, out_b, out_probs, B_, V_, H_, H_, H_, V_, 0, stream);

    // 16. softmax over V
    softmax_rows_kernel<<<B_, 256, 0, stream>>>(out_probs, V_);
}
